// SelfAttention_69114613728420
// MI455X (gfx1250) — matrix operand order repaired and checked by reading, not compiled, not run on hardware
//
#include <hip/hip_runtime.h>
#include <hip/hip_bf16.h>

// MI455X / gfx1250: f16 WMMA (v_wmma_f32_16x16x32_f16) for all three GEMMs,
// fused energy+softmax strip kernel with a 16x4096 f32 tile in LDS (256KB of
// the 320KB WGP LDS), async global->LDS staging of the K fragment, f16 attn
// sidecar so the sa GEMM streams pure b128 loads into WMMA. wave32 throughout.

typedef __attribute__((ext_vector_type(16))) _Float16 v16h;
typedef __attribute__((ext_vector_type(8)))  _Float16 v8h;
typedef __attribute__((ext_vector_type(8)))  float    v8f;
typedef __attribute__((ext_vector_type(4)))  float    f32x4;

#define B_   4
#define C_   256
#define N_   4096
#define CQK  32

// ---- WMMA helpers ---------------------------------------------------------
// A operand (16x32 f16): lane l holds row m = base+(l&15); halves 0..7 are
// K = 8*(l>>4)+0..7, halves 8..15 are K = 16+8*(l>>4)+0..7.
__device__ __forceinline__ v16h ld_a(const _Float16* rowBase, int k0, int half) {
  v16h a;
  ((v8h*)&a)[0] = *(const v8h*)(rowBase + k0 + half * 8);
  ((v8h*)&a)[1] = *(const v8h*)(rowBase + k0 + 16 + half * 8);
  return a;
}
// B operand (32x16 f16): lane l holds col n = base+(l&15); halves 0..7 are
// K = 8*(l>>4)+0..7, halves 8..15 are K = 16+8*(l>>4)+0..7.
__device__ __forceinline__ v16h ld_b(const _Float16* rowBase, int k0, int half) {
  v16h b;
  const _Float16* p = rowBase + k0 + half * 8;
  ((v8h*)&b)[0] = *(const v8h*)(p);
  ((v8h*)&b)[1] = *(const v8h*)(p + 16);
  return b;
}
__device__ __forceinline__ v8f wmma_f16(v16h a, v16h b, v8f c) {
  return __builtin_amdgcn_wmma_f32_16x16x32_f16(
      /*neg_a=*/false, a, /*neg_b=*/false, b,
      /*c_mod=*/(short)0, c, /*reuse_a=*/false, /*reuse_b=*/false);
}

// ---- Kernel 0a: cast + transpose x[B,C,N] -> xt[B,N,C] (f16) --------------
__global__ void prep_x_kernel(const float* __restrict__ x, _Float16* __restrict__ xt) {
  size_t idx = (size_t)blockIdx.x * blockDim.x + threadIdx.x;
  if (idx >= (size_t)B_ * C_ * N_) return;
  int n = (int)(idx % N_);
  int c = (int)((idx / N_) % C_);
  int b = (int)(idx / ((size_t)C_ * N_));
  xt[((size_t)b * N_ + n) * C_ + c] = (_Float16)x[idx];
}

// ---- Kernel 0b: cast weights to f16 ---------------------------------------
__global__ void prep_w_kernel(const float* __restrict__ wq, const float* __restrict__ wk,
                              const float* __restrict__ wv,
                              _Float16* __restrict__ wqh, _Float16* __restrict__ wkh,
                              _Float16* __restrict__ wvh) {
  int i = blockIdx.x * blockDim.x + threadIdx.x;
  if (i < CQK * C_) { wqh[i] = (_Float16)wq[i]; wkh[i] = (_Float16)wk[i]; }
  if (i < C_ * C_)  { wvh[i] = (_Float16)wv[i]; }
}

// ---- Kernel 1: QKV projections (one wave = one 16x16 output tile) ---------
// qt,kt stored [B,N,CQK] (n-major, c fastest) so the energy GEMM reads its
// K-dimension contiguously; vh stored [B,C,N] for the sa GEMM A operand.
__global__ void qkv_kernel(const _Float16* __restrict__ xt,
                           const _Float16* __restrict__ wqh, const float* __restrict__ bq,
                           const _Float16* __restrict__ wkh, const float* __restrict__ bk,
                           const _Float16* __restrict__ wvh, const float* __restrict__ bv,
                           _Float16* __restrict__ qt, _Float16* __restrict__ kt,
                           _Float16* __restrict__ vh) {
  const int lane = threadIdx.x & 31;
  const int half = lane >> 4;
  const int l15  = lane & 15;
  const int nTile = blockIdx.x;   // 0..N/16-1
  const int t     = blockIdx.y;   // 0,1 -> q ; 2,3 -> k ; 4..19 -> v
  const int b     = blockIdx.z;

  const _Float16* w;
  const float* bias;
  int mo, kind;
  if (t < 2)      { w = wqh; bias = bq; mo = t * 16;       kind = 0; }
  else if (t < 4) { w = wkh; bias = bk; mo = (t - 2) * 16; kind = 1; }
  else            { w = wvh; bias = bv; mo = (t - 4) * 16; kind = 2; }

  const int n = nTile * 16 + l15;
  const _Float16* xrow = xt + ((size_t)b * N_ + n) * C_;
  const _Float16* wrow = w + (size_t)(mo + l15) * C_;

  v8f acc = {};
  #pragma unroll
  for (int k0 = 0; k0 < C_; k0 += 32) {
    v16h a  = ld_a(wrow, k0, half);
    v16h bm = ld_b(xrow, k0, half);
    acc = wmma_f16(a, bm, acc);
  }

  const int oBase = mo + 8 * half;
  if (kind < 2) {
    // transposed store: per lane, j=0..7 are contiguous o in qt[n][.]
    _Float16 tmp[8];
    #pragma unroll
    for (int j = 0; j < 8; ++j) tmp[j] = (_Float16)(acc[j] + bias[oBase + j]);
    _Float16* dst = (kind == 0 ? qt : kt) + ((size_t)b * N_ + n) * CQK + oBase;
    *(f32x4*)dst = *(const f32x4*)tmp;
  } else {
    #pragma unroll
    for (int j = 0; j < 8; ++j) {
      int o = oBase + j;
      vh[((size_t)b * C_ + o) * N_ + n] = (_Float16)(acc[j] + bias[o]);
    }
  }
}

// ---- Kernel 2: fused energy + softmax -------------------------------------
// One workgroup owns a 16-row strip of S = K^T Q (rows m, cols n), entirely in
// LDS (16*4096*4 = 256KB dynamic LDS). The 16x32 kt strip (1KB) is staged into
// static LDS via CDNA5 async global->LDS copy (ASYNCcnt). Row-softmax over n
// writes at_map[b][m][n] (f32 output) and an f16 attn sidecar for the sa GEMM.
__global__ void energy_softmax_kernel(const _Float16* __restrict__ qt,
                                      const _Float16* __restrict__ kt,
                                      float* __restrict__ at_map,
                                      _Float16* __restrict__ attn_h) {
  __shared__ _Float16 kstage[16 * CQK];  // 1KB static LDS, before dynamic S[]
  extern __shared__ float S[];           // [16][N_]
  const int tid  = threadIdx.x;
  const int lane = tid & 31;
  const int wave = tid >> 5;          // 0..7
  const int half = lane >> 4;
  const int l15  = lane & 15;
  const int mStrip = blockIdx.x;      // 0..N/16-1
  const int b      = blockIdx.y;

  // Async-stage the 16x32 f16 kt strip (1024B) into LDS: wave 0, two rounds
  // of 32 lanes x 16B via global_load_async_to_lds_b128 (tracked by ASYNCcnt).
  const _Float16* ktStrip = kt + ((size_t)b * N_ + (size_t)mStrip * 16) * CQK;
  if (wave == 0) {
    #pragma unroll
    for (int it = 0; it < 2; ++it) {
      unsigned ldsOff = (unsigned)(size_t)(&kstage[0]) + (unsigned)((it * 32 + lane) * 16);
      const _Float16* g = ktStrip + (it * 32 + lane) * 8;  // 8 halves = 16B
      asm volatile("global_load_async_to_lds_b128 %0, %1, off"
                   :: "v"(ldsOff), "v"(g) : "memory");
    }
    asm volatile("s_wait_asynccnt 0" ::: "memory");
  }
  __syncthreads();

  // A fragment (rows m of K^T, K = full Cqk = 32) from LDS, once per wave.
  v16h a = ld_a(&kstage[(size_t)l15 * CQK], 0, half);

  for (int nT = wave; nT < N_ / 16; nT += 8) {
    const int n = nT * 16 + l15;
    const _Float16* qrow = qt + ((size_t)b * N_ + n) * CQK;
    v16h bm = ld_b(qrow, 0, half);
    v8f acc = {};
    acc = wmma_f16(a, bm, acc);
    #pragma unroll
    for (int j = 0; j < 8; ++j)
      S[(j + 8 * half) * N_ + nT * 16 + l15] = acc[j];
  }
  __syncthreads();

  // Row softmax: each wave handles 2 of the 16 rows.
  #pragma unroll
  for (int rr = 0; rr < 2; ++rr) {
    const int r = wave * 2 + rr;
    float* row = S + r * N_;
    float mx = -3.4e38f;
    for (int i = lane; i < N_; i += 32) mx = fmaxf(mx, row[i]);
    #pragma unroll
    for (int o = 16; o > 0; o >>= 1) mx = fmaxf(mx, __shfl_xor(mx, o, 32));
    float sum = 0.0f;
    for (int i = lane; i < N_; i += 32) {
      float e = __expf(row[i] - mx);
      row[i] = e;
      sum += e;
    }
    #pragma unroll
    for (int o = 16; o > 0; o >>= 1) sum += __shfl_xor(sum, o, 32);
    const float inv = 1.0f / sum;
    const size_t rowOff = ((size_t)b * N_ + (mStrip * 16 + r)) * N_;
    float*    dst = at_map + rowOff;
    _Float16* ah  = attn_h + rowOff;
    for (int i = lane; i < N_; i += 32) {
      float p = row[i] * inv;
      dst[i] = p;                 // f32 output (at_map)
      ah[i]  = (_Float16)p;       // f16 sidecar for the sa GEMM
    }
  }
}

// ---- Kernel 3: sa = V * attn_h^T, epilogue out = x + gamma*sa -------------
// One wave computes a 16c x 128m block (8 sub-tiles reuse the A fragment).
// B operand is attn_h row m (f16, contiguous along reduction n) -> the inner
// loop is pure global_load_b128 + v_wmma, no converts.
__global__ void sa_kernel(const float* __restrict__ x,
                          const _Float16* __restrict__ vh,
                          const _Float16* __restrict__ attn_h,
                          const float* __restrict__ gamma,
                          float* __restrict__ out) {
  const int lane = threadIdx.x & 31;
  const int half = lane >> 4;
  const int l15  = lane & 15;
  const int cTile = blockIdx.x;   // 0..C/16-1
  const int mBlk  = blockIdx.y;   // 0..N/128-1
  const int b     = blockIdx.z;

  const _Float16* vrow = vh + ((size_t)b * C_ + (cTile * 16 + l15)) * N_;

  const _Float16* arow[8];
  #pragma unroll
  for (int s = 0; s < 8; ++s) {
    int m = mBlk * 128 + s * 16 + l15;
    arow[s] = attn_h + ((size_t)b * N_ + m) * N_;
  }

  v8f acc[8] = {};
  for (int k0 = 0; k0 < N_; k0 += 32) {
    __builtin_prefetch(vrow + k0 + 256, 0, 0);  // global_prefetch_b8
    v16h a = ld_a(vrow, k0, half);
    #pragma unroll
    for (int s = 0; s < 8; ++s) {
      v16h bm = ld_b(arow[s], k0, half);
      acc[s] = wmma_f16(a, bm, acc[s]);
    }
  }

  const float g = gamma[0];
  #pragma unroll
  for (int s = 0; s < 8; ++s) {
    #pragma unroll
    for (int j = 0; j < 8; ++j) {
      int cc = cTile * 16 + j + 8 * half;
      int m  = mBlk * 128 + s * 16 + l15;
      size_t idx = ((size_t)b * C_ + cc) * N_ + m;
      out[idx] = x[idx] + g * acc[s][j];
    }
  }
}

// ---- Launch ----------------------------------------------------------------
extern "C" void kernel_launch(void* const* d_in, const int* in_sizes, int n_in,
                              void* d_out, int out_size, void* d_ws, size_t ws_size,
                              hipStream_t stream) {
  const float* x     = (const float*)d_in[0];
  const float* wq    = (const float*)d_in[1];
  const float* bq    = (const float*)d_in[2];
  const float* wk    = (const float*)d_in[3];
  const float* bk    = (const float*)d_in[4];
  const float* wv    = (const float*)d_in[5];
  const float* bv    = (const float*)d_in[6];
  const float* gamma = (const float*)d_in[7];

  float* out    = (float*)d_out;
  float* at_map = out + (size_t)B_ * C_ * N_;  // tuple: (out, at_map) concat

  // Workspace carve-up (~152.5 MB; all regions 16B-aligned).
  char* ws = (char*)d_ws;
  _Float16* xt     = (_Float16*)ws; ws += (size_t)B_ * N_ * C_ * 2;
  _Float16* wqh    = (_Float16*)ws; ws += (size_t)CQK * C_ * 2;
  _Float16* wkh    = (_Float16*)ws; ws += (size_t)CQK * C_ * 2;
  _Float16* wvh    = (_Float16*)ws; ws += (size_t)C_ * C_ * 2;
  _Float16* qt     = (_Float16*)ws; ws += (size_t)B_ * N_ * CQK * 2;
  _Float16* kt     = (_Float16*)ws; ws += (size_t)B_ * N_ * CQK * 2;
  _Float16* vh     = (_Float16*)ws; ws += (size_t)B_ * C_ * N_ * 2;
  _Float16* attn_h = (_Float16*)ws; // B*N*N*2 = 134MB

  prep_x_kernel<<<dim3((B_ * C_ * N_) / 256), 256, 0, stream>>>(x, xt);
  prep_w_kernel<<<dim3((C_ * C_) / 256), 256, 0, stream>>>(wq, wk, wv, wqh, wkh, wvh);

  qkv_kernel<<<dim3(N_ / 16, 20, B_), 32, 0, stream>>>(
      xt, wqh, bq, wkh, bk, wvh, bv, qt, kt, vh);

  // 256KB dynamic LDS per workgroup (fits 320KB WGP LDS; one WG per WGP).
  energy_softmax_kernel<<<dim3(N_ / 16, B_), 256, 16 * N_ * sizeof(float), stream>>>(
      qt, kt, at_map, attn_h);

  sa_kernel<<<dim3(C_ / 16, N_ / 128, B_), 32, 0, stream>>>(x, vh, attn_h, gamma, out);
}
